// Block_Attention_3_48593259987101
// MI455X (gfx1250) — matrix operand order repaired and checked by reading, not compiled, not run on hardware
//
#include <hip/hip_runtime.h>

#define BN_EPS 1e-5f

typedef __attribute__((ext_vector_type(16))) __bf16 v16bf;
typedef __attribute__((ext_vector_type(8)))  __bf16 v8bf;
typedef __attribute__((ext_vector_type(4)))  __bf16 v4bf;
typedef __attribute__((ext_vector_type(8)))  float  v8f;

// ------------------------------------------------------------------
// Kernel 0a: fold conv bias + BN into per-row scale/offset.
// row m in [0,1024): conv = m/256 (k,q,v,d), oc = m%256
// ------------------------------------------------------------------
struct RowParams {
    const float* bias[4];
    const float* g[4];
    const float* beta[4];
    const float* mu[4];
    const float* var[4];
};

__global__ void k_row_params(RowParams rp, float* __restrict__ rowScale,
                             float* __restrict__ rowOff) {
    int m = blockIdx.x * blockDim.x + threadIdx.x;
    if (m >= 1024) return;
    int conv = m >> 8, oc = m & 255;
    float a = rp.g[conv][oc] * rsqrtf(rp.var[conv][oc] + BN_EPS);
    rowScale[m] = a;
    rowOff[m]   = (rp.bias[conv][oc] - rp.mu[conv][oc]) * a + rp.beta[conv][oc];
}

// ------------------------------------------------------------------
// Kernel 0b: passthrough copy x -> out channels [0,2048)
// ------------------------------------------------------------------
__global__ void k_copy_x(const float* __restrict__ x, float* __restrict__ out) {
    int t  = blockIdx.x * blockDim.x + threadIdx.x;   // 0 .. 2^21-1
    int hw = t & 255;
    int c  = (t >> 8) & 2047;
    int b  = t >> 19;
    out[((b * 2304 + c) << 8) + hw] = x[t];
}

// ------------------------------------------------------------------
// Kernel 0c: pack the four weight matrices into A_bf16[m][k], m = conv*256+oc.
// 4 elements / thread, fully coalesced both sides.
// ------------------------------------------------------------------
struct WPtrs { const float* w[4]; };

__global__ __launch_bounds__(256)
void k_pack_w(WPtrs wp, __bf16* __restrict__ A) {
    int t  = blockIdx.x * blockDim.x + threadIdx.x;   // 524288 threads
    int e0 = t * 4;                                   // element base
    int conv = e0 >> 19;                              // 524288 elems per conv
    int off  = e0 & 524287;
    const float4 v = *(const float4*)(wp.w[conv] + off);
    v4bf o;
    o[0] = (__bf16)v.x; o[1] = (__bf16)v.y; o[2] = (__bf16)v.z; o[3] = (__bf16)v.w;
    *(v4bf*)(A + e0) = o;
}

// ------------------------------------------------------------------
// Kernel 0d: transpose+convert x -> Bt_bf16[n][k] (n = b*256+hw, k = channel)
// via 64x64 LDS tile so both global read and write are coalesced.
// grid = (16 n-tiles, 32 k-tiles), 256 threads.
// ------------------------------------------------------------------
__global__ __launch_bounds__(256)
void k_pack_x(const float* __restrict__ x, __bf16* __restrict__ Bt) {
    __shared__ __bf16 tile[64][65];                   // +1 pad vs bank conflicts
    const int n0 = blockIdx.x * 64;
    const int k0 = blockIdx.y * 64;
    const int b   = n0 >> 8;                          // 64 | 256: tile in one batch
    const int hw0 = n0 & 255;
    const int t = threadIdx.x;
    {
        const int tn = t & 63, tk4 = t >> 6;          // read: n contiguous
        for (int i = 0; i < 16; i++) {
            int kl = i * 4 + tk4;
            tile[kl][tn] =
                (__bf16)x[(size_t)b * 524288 + (size_t)(k0 + kl) * 256 + hw0 + tn];
        }
    }
    __syncthreads();
    {
        const int tk = t & 63, tn4 = t >> 6;          // write: k contiguous
        for (int i = 0; i < 16; i++) {
            int nl = i * 4 + tn4;
            Bt[(size_t)(n0 + nl) * 2048 + k0 + tk] = tile[tk][nl];
        }
    }
}

// ------------------------------------------------------------------
// Kernel 1: fused K/Q/V/delta GEMM, bf16 WMMA, f32 accumulate.
//   feat(1024 x 1024) = A(1024 x 2048) * Bt^T, epilogue = BN fold (+ReLU m>=768)
// Block: 256 threads (8 waves), 128x128 tile, K-step 32, register-pipelined.
// ------------------------------------------------------------------
__global__ __launch_bounds__(256)
void k_gemm_kqvd(const __bf16* __restrict__ A, const __bf16* __restrict__ Bt,
                 const float* __restrict__ rowScale,
                 const float* __restrict__ rowOff,
                 float* __restrict__ feat) {
    __shared__ __bf16 A_s[128 * 32];   // [m][k], k contiguous
    __shared__ __bf16 B_s[128 * 32];   // [n][k], k contiguous

    const int t     = threadIdx.x;
    const int lane  = t & 31;
    const int wid   = t >> 5;
    const int lrow  = lane & 15;
    const int lhalf = lane >> 4;
    const int waveM = wid & 3;   // 4 waves along M
    const int waveN = wid >> 2;  // 2 waves along N

    const int m0 = blockIdx.y * 128;
    const int n0 = blockIdx.x * 128;

    // staging: thread t owns row sr (of both tiles), k-half sh
    const int sr = t >> 1, sh = (t & 1) * 16;
    const __bf16* ga = A  + (size_t)(m0 + sr) * 2048 + sh;
    const __bf16* gb = Bt + (size_t)(n0 + sr) * 2048 + sh;

    v8f acc[2][4];
#pragma unroll
    for (int i = 0; i < 2; i++)
#pragma unroll
        for (int j = 0; j < 4; j++) { v8f z = {}; acc[i][j] = z; }

    // prologue: tile 0 into registers
    v16bf ra = *(const v16bf*)ga;
    v16bf rb = *(const v16bf*)gb;

    for (int k0 = 0; k0 < 2048; k0 += 32) {
        *(v16bf*)&A_s[sr * 32 + sh] = ra;
        *(v16bf*)&B_s[sr * 32 + sh] = rb;
        __syncthreads();
        if (k0 + 32 < 2048) {             // overlap next tile's loads with WMMAs
            ra = *(const v16bf*)(ga + k0 + 32);
            rb = *(const v16bf*)(gb + k0 + 32);
        }

        // A fragments (16x32 bf16): lane L: m=L%16, k = (L/16)*16 .. +15
        v16bf afrag[2];
#pragma unroll
        for (int tm = 0; tm < 2; tm++) {
            const __bf16* ap = &A_s[(waveM * 32 + tm * 16 + lrow) * 32 + lhalf * 16];
            v8bf lo = *(const v8bf*)ap;
            v8bf hi = *(const v8bf*)(ap + 8);
#pragma unroll
            for (int i = 0; i < 8; i++) { afrag[tm][i] = lo[i]; afrag[tm][i + 8] = hi[i]; }
        }
        // B fragments (32x16 bf16): lane L: n=L%16, k = (L/16)*16 .. +15
#pragma unroll
        for (int tn = 0; tn < 4; tn++) {
            const __bf16* bp = &B_s[(waveN * 64 + tn * 16 + lrow) * 32 + lhalf * 16];
            v16bf bfrag = *(const v16bf*)bp;
#pragma unroll
            for (int tm = 0; tm < 2; tm++) {
                acc[tm][tn] = __builtin_amdgcn_wmma_f32_16x16x32_bf16(
                    false, afrag[tm], false, bfrag, (short)0, acc[tm][tn],
                    false, false);
            }
        }
        __syncthreads();
    }

    // epilogue: BN fold, ReLU for delta rows, store feat[m][n]
    // C layout: lane L -> n = L%16; VGPR r -> m = r + 8*(L/16)
#pragma unroll
    for (int tm = 0; tm < 2; tm++) {
#pragma unroll
        for (int tn = 0; tn < 4; tn++) {
            const int mBase = m0 + waveM * 32 + tm * 16 + lhalf * 8;
            const int n     = n0 + waveN * 64 + tn * 16 + lrow;
#pragma unroll
            for (int r = 0; r < 8; r++) {
                int m = mBase + r;
                float y = acc[tm][tn][r] * rowScale[m] + rowOff[m];
                if (m >= 768) y = fmaxf(y, 0.f);   // inc = relu(bn(conv_d))
                feat[m * 1024 + n] = y;
            }
        }
    }
}

// ------------------------------------------------------------------
// Kernel 2: per-pixel channel softmax of inc, then S[b,c,h] = sum_w incx.
// One block per (b,h), thread = c.
// ------------------------------------------------------------------
__global__ __launch_bounds__(256)
void k_softmax_S(const float* __restrict__ feat, float* __restrict__ S) {
    __shared__ float vals[16 * 256];
    __shared__ float mx_s[16], sum_s[16];
    const int c  = threadIdx.x;
    const int bh = blockIdx.x;          // b*16 + h
    const int b  = bh >> 4, h = bh & 15;
    const float* row = feat + (768 + c) * 1024 + b * 256 + h * 16;
    float v[16];
#pragma unroll
    for (int w = 0; w < 16; w++) { v[w] = row[w]; vals[w * 256 + c] = v[w]; }
    __syncthreads();
    if (c < 16) {                        // per-pixel (w = c) channel reduction
        float mx = -1e30f;
        for (int i = 0; i < 256; i++) mx = fmaxf(mx, vals[c * 256 + i]);
        float s = 0.f;
        for (int i = 0; i < 256; i++) s += expf(vals[c * 256 + i] - mx);
        mx_s[c] = mx; sum_s[c] = s;
    }
    __syncthreads();
    float acc = 0.f;
#pragma unroll
    for (int w = 0; w < 16; w++) acc += expf(v[w] - mx_s[w]) / sum_s[w];
    S[(b * 256 + c) * 16 + h] = acc;
}

// ------------------------------------------------------------------
// Kernel 3: per-(batch,patch) attention + output BN.
// One block per (b,p), 256 threads (= channel c, reused as (n,m) pairs).
// ------------------------------------------------------------------
__global__ __launch_bounds__(256)
void k_attention(const float* __restrict__ feat, const float* __restrict__ S,
                 const float* __restrict__ pos,
                 const float* __restrict__ go, const float* __restrict__ beto,
                 const float* __restrict__ mo, const float* __restrict__ vo,
                 float* __restrict__ out) {
    __shared__ float J_s[256 * 16];
    __shared__ float K_s[256 * 16];
    __shared__ float V_s[256 * 16];
    __shared__ float sc_s[16 * 16];
    __shared__ float att_s[16 * 16];

    const int c  = threadIdx.x;
    const int bp = blockIdx.x;       // b*16 + p
    const int b  = bp >> 4, p = bp & 15;
    const int pi = p >> 2, pj = p & 3;

    const float sval = S[(b * 256 + c) * 16 + p];  // S_up constant per patch
#pragma unroll
    for (int n = 0; n < 16; n++) {
        int ph = n >> 2, pw = n & 3;
        int h = pi * 4 + ph, w = pj * 4 + pw;
        int g = b * 256 + h * 16 + w;
        float pv = pos[b * 65536 + c * 256 + h * 16 + w];
        K_s[c * 16 + n] = feat[c * 1024 + g] + pv;
        J_s[c * 16 + n] = feat[(256 + c) * 1024 + g] * sval + pv;
        V_s[c * 16 + n] = feat[(512 + c) * 1024 + g] + pv;
    }
    __syncthreads();

    // scores[n][m] = sum_c J[c][n]*K[c][m]; thread = (n,m)
    {
        int n = c >> 4, m = c & 15;
        float s = 0.f;
        for (int ch = 0; ch < 256; ch++)
            s += J_s[ch * 16 + n] * K_s[ch * 16 + m];
        sc_s[n * 16 + m] = s;
    }
    __syncthreads();
    // softmax over n (queries) per key column m  (reference dim=2)
    {
        int n = c >> 4, m = c & 15;
        float mx = -1e30f;
#pragma unroll
        for (int q = 0; q < 16; q++) mx = fmaxf(mx, sc_s[q * 16 + m]);
        float sum = 0.f;
#pragma unroll
        for (int q = 0; q < 16; q++) sum += expf(sc_s[q * 16 + m] - mx);
        att_s[n * 16 + m] = expf(sc_s[n * 16 + m] - mx) / sum;
    }
    __syncthreads();

    // out[n][c] = sum_m att[n][m] * V[c][m]; output BN; scatter to image
    float alpha = go[c] * rsqrtf(vo[c] + BN_EPS);
    float mu = mo[c], bet = beto[c];
#pragma unroll
    for (int n = 0; n < 16; n++) {
        float o = 0.f;
#pragma unroll
        for (int m = 0; m < 16; m++) o += att_s[n * 16 + m] * V_s[c * 16 + m];
        float y = (o - mu) * alpha + bet;
        int ph = n >> 2, pw = n & 3;
        int h = pi * 4 + ph, w = pj * 4 + pw;
        out[((b * 2304 + 2048 + c) << 8) + h * 16 + w] = y;
    }
}

// ------------------------------------------------------------------
extern "C" void kernel_launch(void* const* d_in, const int* in_sizes, int n_in,
                              void* d_out, int out_size, void* d_ws, size_t ws_size,
                              hipStream_t stream) {
    const float* x    = (const float*)d_in[0];
    const float* go   = (const float*)d_in[25];
    const float* beto = (const float*)d_in[26];
    const float* mo   = (const float*)d_in[27];
    const float* vo   = (const float*)d_in[28];
    const float* pos  = (const float*)d_in[29];

    RowParams rp;
    WPtrs wp;
    const int base[4] = {1, 7, 13, 19};   // w, b, g, beta, m, v per conv
    for (int i = 0; i < 4; i++) {
        wp.w[i]    = (const float*)d_in[base[i]];
        rp.bias[i] = (const float*)d_in[base[i] + 1];
        rp.g[i]    = (const float*)d_in[base[i] + 2];
        rp.beta[i] = (const float*)d_in[base[i] + 3];
        rp.mu[i]   = (const float*)d_in[base[i] + 4];
        rp.var[i]  = (const float*)d_in[base[i] + 5];
    }

    char* wsB = (char*)d_ws;
    float*  rowScale = (float*)wsB;                         // 1024 f32
    float*  rowOff   = rowScale + 1024;                     // 1024 f32
    float*  feat     = rowOff + 1024;                       // 1024*1024 f32
    float*  S        = feat + 1024 * 1024;                  // 16384 f32
    __bf16* Abf      = (__bf16*)(S + 16384);                // 1024*2048 bf16
    __bf16* Btbf     = Abf + 1024 * 2048;                   // 1024*2048 bf16
    float*  outf     = (float*)d_out;

    k_row_params<<<4, 256, 0, stream>>>(rp, rowScale, rowOff);
    k_copy_x<<<8192, 256, 0, stream>>>(x, outf);
    k_pack_w<<<2048, 256, 0, stream>>>(wp, Abf);
    dim3 gt(16, 32);
    k_pack_x<<<gt, 256, 0, stream>>>(x, Btbf);
    dim3 g1(8, 8);   // N tiles x M tiles
    k_gemm_kqvd<<<g1, 256, 0, stream>>>(Abf, Btbf, rowScale, rowOff, feat);
    k_softmax_S<<<64, 256, 0, stream>>>(feat, S);
    k_attention<<<64, 256, 0, stream>>>(feat, S, pos, go, beto, mo, vo, outf);
}
